// GQA_57406532878374
// MI455X (gfx1250) — matrix operand order repaired and checked by reading, not compiled, not run on hardware
//
#include <hip/hip_runtime.h>
#include <hip/hip_bf16.h>

// ---------------- problem constants ----------------
#define BB   4
#define SS   2048
#define DIM  1152
#define NH   12
#define HD   96
#define KVH  2
#define GRP  6
#define MROWS (BB*SS)           // 8192 rows for the projections
#define SCALE 0.10206207261596575f  // 96^-0.5

// ---------------- WMMA types ----------------
typedef __attribute__((ext_vector_type(16))) __bf16 v16bf;
typedef __attribute__((ext_vector_type(8)))  float  v8f;
typedef unsigned int u32x4 __attribute__((ext_vector_type(4)));
typedef int          i32x8 __attribute__((ext_vector_type(8)));
typedef int          i32x4 __attribute__((ext_vector_type(4)));

union ABReg { unsigned u[8]; v16bf v; };

static __device__ __forceinline__ unsigned short f2bf(float f) {
    unsigned u = __float_as_uint(f);
    unsigned r = u + 0x7FFFu + ((u >> 16) & 1u);   // round-to-nearest-even
    return (unsigned short)(r >> 16);
}

// A-matrix (16x32 bf16) K index for VGPR v, half-wave hg  (ISA 7.12.2)
static __device__ __forceinline__ int a_kidx(int v, int hg) {
    return (v < 4) ? (2 * v + 8 * hg) : (16 + 2 * (v - 4) + 8 * hg);
}
// B-matrix (32x16 bf16) K index for VGPR v, half-wave hg
static __device__ __forceinline__ int b_kidx(int v, int hg) {
    return (v < 4) ? (2 * v + 8 * hg) : (16 + 2 * (v - 4) + 8 * hg);
}

#define WMMA_BF16(A, Bm, C) \
    __builtin_amdgcn_wmma_f32_16x16x32_bf16(false, (A), false, (Bm), (short)0, (C), false, false)

// ---- TDM: DMA a 32x96 bf16 tile (row stride 192 elements) from global into LDS ----
// D# per CDNA5 ISA: group0 = {flags, lds_addr, global_addr, type=2},
// group1 = {mask/data_size, tensor_dim0/1, tile_dim0/1/2, tensor_dim0_stride}.
static __device__ __forceinline__ void tdm_load_tile_32x96(const unsigned short* gsrc,
                                                           unsigned lds_off) {
    unsigned long long ga = (unsigned long long)(size_t)gsrc;
    u32x4 g0;
    g0[0] = 1u;                                            // count=1 (valid user descriptor)
    g0[1] = lds_off;                                       // lds_addr (bytes)
    g0[2] = (unsigned)(ga & 0xFFFFFFFFu);                  // global_addr[31:0]
    g0[3] = (unsigned)((ga >> 32) & 0x01FFFFFFu) | (2u << 30); // addr[56:32] | type=2
    i32x8 g1;
    g1[0] = (1 << 16);          // workgroup_mask=0, data_size=1 (2-byte elements)
    g1[1] = (int)(96u << 16);   // tensor_dim0 = 96   (bits 79:48, low half)
    g1[2] = (int)(32u << 16);   // tensor_dim1 = 32   (bits 111:80, low half)
    g1[3] = (int)(96u << 16);   // tile_dim0  = 96    (bits 127:112)
    g1[4] = 32;                 // tile_dim1  = 32    (bits 143:128); tile_dim2 = 0
    g1[5] = 192;                // tensor_dim0_stride = 192 (bits 207:160, low 32)
    g1[6] = 0;
    g1[7] = 0;
    i32x4 z4 = {0, 0, 0, 0};            // groups 2/3 unused (2D tensor)
    i32x8 z8 = {0, 0, 0, 0, 0, 0, 0, 0};
    __builtin_amdgcn_tensor_load_to_lds(g0, g1, z4, z4, z8, 0);
}

// ---------------- fp32 -> bf16 convert ----------------
__global__ void cvt_bf16_kernel(const float* __restrict__ in,
                                unsigned short* __restrict__ out, int n) {
    int i = blockIdx.x * blockDim.x + threadIdx.x;
    if (i < n) out[i] = f2bf(in[i]);
}

// ---------------- generic WMMA GEMM:  Y[M,N] = Xb[M,1152] * Wb[N,1152]^T + bias ----------------
__global__ void gemm_bf16_kernel(const unsigned short* __restrict__ Xb,
                                 const unsigned short* __restrict__ Wb,
                                 const float* __restrict__ bias,
                                 float* __restrict__ Y,
                                 int Ncols, int ngroups) {
    const int lane = threadIdx.x & 31;
    const int w    = blockIdx.x * (blockDim.x >> 5) + (threadIdx.x >> 5);
    const int mtile = w / ngroups;
    const int ng    = w % ngroups;
    const int n0    = ng * 64;
    const int hg = lane >> 4, ln = lane & 15;
    const size_t row = (size_t)mtile * 16 + ln;

    v8f acc[4] = {v8f{}, v8f{}, v8f{}, v8f{}};

    for (int k0 = 0; k0 < DIM; k0 += 32) {
        ABReg a;
#pragma unroll
        for (int v = 0; v < 8; ++v)
            a.u[v] = *(const unsigned*)(Xb + row * DIM + k0 + a_kidx(v, hg));
#pragma unroll
        for (int j = 0; j < 4; ++j) {
            ABReg bm;
            const size_t ncol = n0 + j * 16 + ln;
#pragma unroll
            for (int v = 0; v < 8; ++v)
                bm.u[v] = *(const unsigned*)(Wb + ncol * DIM + k0 + b_kidx(v, hg));
            acc[j] = WMMA_BF16(a.v, bm.v, acc[j]);
        }
    }
#pragma unroll
    for (int j = 0; j < 4; ++j) {
#pragma unroll
        for (int v = 0; v < 8; ++v) {
            const int m   = hg * 8 + v;
            const int col = n0 + j * 16 + ln;
            Y[((size_t)mtile * 16 + m) * Ncols + col] = acc[j][v] + bias[col];
        }
    }
}

// ---------------- RoPE on q (fp32 in, bf16 out) ----------------
__global__ void rope_q_kernel(const float* __restrict__ qf,
                              unsigned short* __restrict__ rq) {
    int idx = blockIdx.x * blockDim.x + threadIdx.x;
    if (idx >= BB * SS * NH * (HD / 2)) return;
    int p = idx % (HD / 2); int t = idx / (HD / 2);
    int h = t % NH; t /= NH;
    int s = t % SS; int b = t / SS;
    const size_t base = ((size_t)(b * SS + s) * NH + h) * HD;
    float e = qf[base + 2 * p], o = qf[base + 2 * p + 1];
    float freq = __expf(-(2.0f * p / (float)HD) * 9.210340371976184f);
    float ang = (float)s * freq, sn, cs;
    __sincosf(ang, &sn, &cs);
    rq[base + 2 * p]     = f2bf(e * cs - o * sn);
    rq[base + 2 * p + 1] = f2bf(o * cs + e * sn);
}

// ---------------- RoPE on k + bf16 convert of v ----------------
__global__ void rope_kv_kernel(const float* __restrict__ kvf,
                               unsigned short* __restrict__ rk,
                               unsigned short* __restrict__ vb) {
    int idx = blockIdx.x * blockDim.x + threadIdx.x;
    if (idx >= BB * SS * KVH * (HD / 2)) return;
    int p = idx % (HD / 2); int t = idx / (HD / 2);
    int kvh = t % KVH; t /= KVH;
    int s = t % SS; int b = t / SS;
    const size_t kbase = (((size_t)(b * SS + s) * 2 + 0) * KVH + kvh) * HD;
    const size_t vbase = (((size_t)(b * SS + s) * 2 + 1) * KVH + kvh) * HD;
    const size_t obase = ((size_t)(b * SS + s) * KVH + kvh) * HD;
    float e = kvf[kbase + 2 * p], o = kvf[kbase + 2 * p + 1];
    float freq = __expf(-(2.0f * p / (float)HD) * 9.210340371976184f);
    float ang = (float)s * freq, sn, cs;
    __sincosf(ang, &sn, &cs);
    rk[obase + 2 * p]     = f2bf(e * cs - o * sn);
    rk[obase + 2 * p + 1] = f2bf(o * cs + e * sn);
    vb[obase + 2 * p]     = f2bf(kvf[vbase + 2 * p]);
    vb[obase + 2 * p + 1] = f2bf(kvf[vbase + 2 * p + 1]);
}

// ---------------- flash attention ----------------
// one block = 4 waves sharing (b,h); wave wl handles q-tile (qg*4 + wl).
// Double-buffered TDM pipeline: at tile i, wave 0 issues tile i+1's K/V DMAs
// into the other LDS buffer, then waits TENSORcnt<=2 (tile i's pair done,
// tile i+1's pair still in flight) so DMA overlaps WMMA compute.
__global__ void attn_kernel(const unsigned short* __restrict__ rq,
                            const unsigned short* __restrict__ rk,
                            const unsigned short* __restrict__ vb,
                            const float* __restrict__ mask,
                            const float* __restrict__ pad,
                            unsigned short* __restrict__ aout) {
    __shared__ unsigned short ksh[2][32 * 96];    // K tiles, row-major [token][dim]
    __shared__ unsigned short vsh[2][32 * 96];    // V tiles, row-major [token][dim]
    __shared__ unsigned short pbuf[4][16 * 32];   // per-wave P staging

    const int lane = threadIdx.x & 31;
    const int wl   = threadIdx.x >> 5;
    const int bh   = blockIdx.x >> 5;             // 48 (b,h) pairs
    const int qg   = blockIdx.x & 31;             // 32 q-groups of 4 tiles
    const int h = bh % NH, b = bh / NH;
    const int kvh = h / GRP;
    const int hg = lane >> 4, ln = lane & 15;
    const int q0 = (qg * 4 + wl) * 16;

    const unsigned ks_off[2] = { (unsigned)(size_t)(&ksh[0][0]),
                                 (unsigned)(size_t)(&ksh[1][0]) };
    const unsigned vs_off[2] = { (unsigned)(size_t)(&vsh[0][0]),
                                 (unsigned)(size_t)(&vsh[1][0]) };
    const unsigned short* kv_base_k = rk + ((size_t)b * SS * KVH + kvh) * HD;
    const unsigned short* kv_base_v = vb + ((size_t)b * SS * KVH + kvh) * HD;

    // load Q tile (16x96) into 3 A operands, kept in registers
    ABReg aq[3];
    {
        const unsigned short* qrow = rq + ((size_t)(b * SS + q0 + ln) * NH + h) * HD;
#pragma unroll
        for (int c = 0; c < 3; ++c)
#pragma unroll
            for (int v = 0; v < 8; ++v)
                aq[c].u[v] = *(const unsigned*)(qrow + c * 32 + a_kidx(v, hg));
    }

    v8f o[6]; float mrow[8], lrow[8];
#pragma unroll
    for (int j = 0; j < 6; ++j) o[j] = v8f{};
#pragma unroll
    for (int v = 0; v < 8; ++v) { mrow[v] = -1e30f; lrow[v] = 0.0f; }

    const int NT = SS / 32;                       // 64 key tiles
    if (wl == 0) {                                // prologue: stage tile 0 into buffer 0
        tdm_load_tile_32x96(kv_base_k, ks_off[0]);
        tdm_load_tile_32x96(kv_base_v, vs_off[0]);
    }

    for (int it = 0; it < NT; ++it) {
        const int t0  = it * 32;
        const int buf = it & 1;
        if (wl == 0) {
            if (it + 1 < NT) {                    // issue next tile into other buffer
                const size_t goff = (size_t)(t0 + 32) * KVH * HD;
                tdm_load_tile_32x96(kv_base_k + goff, ks_off[buf ^ 1]);
                tdm_load_tile_32x96(kv_base_v + goff, vs_off[buf ^ 1]);
                __builtin_amdgcn_s_wait_tensorcnt(2);   // oldest pair (tile it) complete
            } else {
                __builtin_amdgcn_s_wait_tensorcnt(0);   // last tile: drain
            }
        }
        __syncthreads();

        const unsigned short* kcur = &ksh[buf][0];
        const unsigned short* vcur = &vsh[buf][0];

        // ---- S = Q * K^T for 2 key tiles (t0..t0+15, t0+16..t0+31) ----
        v8f s0 = v8f{}, s1 = v8f{};
#pragma unroll
        for (int c = 0; c < 3; ++c) {
            ABReg bk0, bk1;
#pragma unroll
            for (int v = 0; v < 8; ++v) {
                const int kk = c * 32 + b_kidx(v, hg);
                bk0.u[v] = *(const unsigned*)(kcur + ln * 96 + kk);
                bk1.u[v] = *(const unsigned*)(kcur + (16 + ln) * 96 + kk);
            }
            s0 = WMMA_BF16(aq[c].v, bk0.v, s0);
            s1 = WMMA_BF16(aq[c].v, bk1.v, s1);
        }
        // ---- scale + mask + pad, online softmax ----
        float p0[8], p1[8], mt[8];
#pragma unroll
        for (int v = 0; v < 8; ++v) {
            const int row = q0 + hg * 8 + v;
            const int c0 = t0 + ln, c1 = t0 + 16 + ln;
            p0[v] = s0[v] * SCALE + mask[(size_t)row * SS + c0] + pad[(size_t)b * SS + c0];
            p1[v] = s1[v] * SCALE + mask[(size_t)row * SS + c1] + pad[(size_t)b * SS + c1];
            mt[v] = fmaxf(p0[v], p1[v]);
        }
#pragma unroll
        for (int off = 8; off > 0; off >>= 1)          // row max across the 16-lane half
#pragma unroll
            for (int v = 0; v < 8; ++v)
                mt[v] = fmaxf(mt[v], __shfl_xor(mt[v], off, 32));
#pragma unroll
        for (int v = 0; v < 8; ++v) {
            const float mn = fmaxf(mrow[v], mt[v]);
            const float corr = __expf(mrow[v] - mn);
            mrow[v] = mn; lrow[v] *= corr;
#pragma unroll
            for (int j = 0; j < 6; ++j) o[j][v] *= corr;
            p0[v] = __expf(p0[v] - mn);
            p1[v] = __expf(p1[v] - mn);
        }
        float rs[8];
#pragma unroll
        for (int v = 0; v < 8; ++v) rs[v] = p0[v] + p1[v];
#pragma unroll
        for (int off = 8; off > 0; off >>= 1)
#pragma unroll
            for (int v = 0; v < 8; ++v)
                rs[v] += __shfl_xor(rs[v], off, 32);
#pragma unroll
        for (int v = 0; v < 8; ++v) lrow[v] += rs[v];

        // ---- P (C layout) -> LDS -> A layout bf16 ----
        unsigned short* pb = pbuf[wl];
#pragma unroll
        for (int v = 0; v < 8; ++v) {
            const int m = hg * 8 + v;
            pb[m * 32 + ln]      = f2bf(p0[v]);
            pb[m * 32 + 16 + ln] = f2bf(p1[v]);
        }
        ABReg ap;
#pragma unroll
        for (int v = 0; v < 8; ++v)
            ap.u[v] = *(const unsigned*)(pb + (size_t)ln * 32 + a_kidx(v, hg));

        // ---- O += P * V  (V B-operands served from LDS) ----
#pragma unroll
        for (int j = 0; j < 6; ++j) {
            ABReg bvr;
#pragma unroll
            for (int v = 0; v < 8; ++v) {
                const int kk = b_kidx(v, hg);      // token offset within the 32-tile
                const unsigned lo = vcur[(size_t)kk * 96 + j * 16 + ln];
                const unsigned hi = vcur[(size_t)(kk + 1) * 96 + j * 16 + ln];
                bvr.u[v] = lo | (hi << 16);
            }
            o[j] = WMMA_BF16(ap.v, bvr.v, o[j]);
        }
        __syncthreads();   // all waves done with this buffer before its next DMA
    }

    // ---- normalize, write bf16 attention output (b, s, DIM) ----
#pragma unroll
    for (int v = 0; v < 8; ++v) {
        const float inv = 1.0f / lrow[v];
        const int m = hg * 8 + v;
        unsigned short* orow = aout + (size_t)(b * SS + q0 + m) * DIM + h * HD;
#pragma unroll
        for (int j = 0; j < 6; ++j)
            orow[j * 16 + ln] = f2bf(o[j][v] * inv);
    }
}

// ---------------- host-side launch ----------------
static inline size_t align256(size_t x) { return (x + 255) & ~(size_t)255; }

extern "C" void kernel_launch(void* const* d_in, const int* in_sizes, int n_in,
                              void* d_out, int out_size, void* d_ws, size_t ws_size,
                              hipStream_t stream) {
    const float* x     = (const float*)d_in[0];
    const float* mask  = (const float*)d_in[1];
    const float* pad   = (const float*)d_in[2];
    const float* q_w   = (const float*)d_in[3];
    const float* q_b   = (const float*)d_in[4];
    const float* kv_w  = (const float*)d_in[5];
    const float* kv_b  = (const float*)d_in[6];
    const float* wo_w  = (const float*)d_in[7];
    const float* wo_b  = (const float*)d_in[8];
    float* out = (float*)d_out;

    char* p = (char*)d_ws;
    unsigned short* xb    = (unsigned short*)p; p += align256((size_t)MROWS * DIM * 2);
    unsigned short* qwb   = (unsigned short*)p; p += align256((size_t)DIM * DIM * 2);
    unsigned short* kvwb  = (unsigned short*)p; p += align256((size_t)(2 * KVH * HD) * DIM * 2);
    unsigned short* wowb  = (unsigned short*)p; p += align256((size_t)DIM * DIM * 2);
    float*          qf    = (float*)p;          p += align256((size_t)MROWS * DIM * 4);
    float*          kvf   = (float*)p;          p += align256((size_t)MROWS * (2 * KVH * HD) * 4);
    unsigned short* rq    = (unsigned short*)p; p += align256((size_t)MROWS * DIM * 2);
    unsigned short* rk    = (unsigned short*)p; p += align256((size_t)MROWS * KVH * HD * 2);
    unsigned short* vbuf  = (unsigned short*)p; p += align256((size_t)MROWS * KVH * HD * 2);
    unsigned short* attnb = (unsigned short*)p; p += align256((size_t)MROWS * DIM * 2);

    // 1) fp32 -> bf16 conversions
    {
        int n = MROWS * DIM;
        cvt_bf16_kernel<<<(n + 255) / 256, 256, 0, stream>>>(x, xb, n);
        n = DIM * DIM;
        cvt_bf16_kernel<<<(n + 255) / 256, 256, 0, stream>>>(q_w, qwb, n);
        n = 2 * KVH * HD * DIM;
        cvt_bf16_kernel<<<(n + 255) / 256, 256, 0, stream>>>(kv_w, kvwb, n);
        n = DIM * DIM;
        cvt_bf16_kernel<<<(n + 255) / 256, 256, 0, stream>>>(wo_w, wowb, n);
    }

    // 2) projections (bf16 WMMA, fp32 accumulate)
    {
        const int mtiles = MROWS / 16;                  // 512
        int ng = DIM / 64;                              // 18
        gemm_bf16_kernel<<<mtiles * ng / 4, 128, 0, stream>>>(xb, qwb, q_b, qf, DIM, ng);
        ng = (2 * KVH * HD) / 64;                       // 6
        gemm_bf16_kernel<<<mtiles * ng / 4, 128, 0, stream>>>(xb, kvwb, kv_b, kvf, 2 * KVH * HD, ng);
    }

    // 3) RoPE + v conversion
    {
        int n = BB * SS * NH * (HD / 2);
        rope_q_kernel<<<(n + 255) / 256, 256, 0, stream>>>(qf, rq);
        n = BB * SS * KVH * (HD / 2);
        rope_kv_kernel<<<(n + 255) / 256, 256, 0, stream>>>(kvf, rk, vbuf);
    }

    // 4) flash attention (double-buffered TDM-staged K/V tiles)
    {
        const int blocks = BB * NH * (SS / 16) / 4;     // 1536
        attn_kernel<<<blocks, 128, 0, stream>>>(rq, rk, vbuf, mask, pad, attnb);
    }

    // 5) output projection -> fp32 d_out
    {
        const int mtiles = MROWS / 16;
        const int ng = DIM / 64;
        gemm_bf16_kernel<<<mtiles * ng / 4, 128, 0, stream>>>(attnb, wowb, wo_b, out, DIM, ng);
    }
}
